// PPRGo_mag_6519760355654
// MI455X (gfx1250) — matrix operand order repaired and checked by reading, not compiled, not run on hardware
//
#include <hip/hip_runtime.h>
#include <hip/hip_bf16.h>
#include <math.h>

// ---------------------------------------------------------------------------
// PPRGo fused forward for MI455X (gfx1250, wave32, WMMA bf16 + async-LDS)
//   gather -> [64x128]@[128x512] relu -> @[512x512] relu -> @[512x192(pad)]
//   -> weighted K-aggregation -> log_softmax
// Weights are pre-packed to bf16 in a contiguous-k (k = 16h + s) V_WMMA_F32_16X16X32_BF16
// B-fragment layout. Each workgroup stages weight slices into LDS ONCE via
// global_load_async_to_lds_b128 (double-buffered, ASYNCcnt-synchronized), so
// all 8 waves share them through LDS instead of 8x-rereading L2.
// ---------------------------------------------------------------------------

typedef __attribute__((ext_vector_type(8)))  __bf16 v8bf;
typedef __attribute__((ext_vector_type(16))) __bf16 v16bf;
typedef __attribute__((ext_vector_type(8)))  float  v8f;

#define NUM_FEATURES 128
#define HIDDEN       512
#define NUM_CLASSES  172
#define NPAD         192   // 12 tiles of 16
#define KNBR         32
#define MTILE        64    // rows per workgroup = 2 batch rows * K

// LDS map (bytes):
//  region0 @0      : 64KB  sH1 bf16 64x512  -> later sLogits f32 64x192 (48KB)
//  region1 @65536  : 64KB  sA  bf16 64x128 (16KB) -> later sH2 bf16 64x512
//  sB0     @131072 : 64KB  weight staging buffer 0
//  sB1     @196608 : 64KB  weight staging buffer 1
//  small   @262144 : 2KB   gidx[64] int, w_s[64] float, sAgg[2*192] float
#define LDS_REGION0_OFF 0
#define LDS_REGION1_OFF 65536
#define LDS_SB0_OFF     131072
#define LDS_SB1_OFF     196608
#define LDS_SMALL_OFF   262144
#define LDS_TOTAL       264192

__device__ __forceinline__ unsigned short f2bf(float f) {
  unsigned u = __builtin_bit_cast(unsigned, f);
  unsigned r = u + 0x7FFFu + ((u >> 16) & 1u);
  return (unsigned short)(r >> 16);
}
__device__ __forceinline__ unsigned pack2bf(float lo, float hi) {
  return (unsigned)f2bf(lo) | ((unsigned)f2bf(hi) << 16);
}

// ---- CDNA5 async Global->LDS copy (16B per lane), tracked by ASYNCcnt ----
__device__ __forceinline__ void async_cp16(unsigned ldsoff, const void* src) {
  asm volatile("global_load_async_to_lds_b128 %0, %1, off"
               :: "v"(ldsoff), "v"((unsigned long long)(uintptr_t)src)
               : "memory");
}
template <int N>
__device__ __forceinline__ void wait_async() {
#if __has_builtin(__builtin_amdgcn_s_wait_asynccnt)
  __builtin_amdgcn_s_wait_asynccnt((unsigned short)N);
#else
  asm volatile("s_wait_asynccnt %0" :: "n"(N) : "memory");
#endif
}

// Pack f32 row-major W[Kdim x Nreal] into bf16 B-fragment layout:
// out[((kt*NT + nt)*32 + lane)*16 + t] = W[kt*32 + (lane>>4)*16 + t][nt*16 + (lane&15)]
__global__ void pprgo_pack_w(const float* __restrict__ W, unsigned short* __restrict__ out,
                             int Kdim, int Nreal, int KT, int NT) {
  int total = KT * NT * 32 * 16;
  for (int p = blockIdx.x * blockDim.x + threadIdx.x; p < total; p += gridDim.x * blockDim.x) {
    int t    = p & 15;
    int lane = (p >> 4) & 31;
    int nt   = (p >> 9) % NT;
    int kt   = (p >> 9) / NT;
    int k = kt * 32 + (lane >> 4) * 16 + t;
    int n = nt * 16 + (lane & 15);
    float v = (k < Kdim && n < Nreal) ? W[(size_t)k * Nreal + n] : 0.0f;
    out[p] = f2bf(v);
  }
}

// Stage one 4-ntile chunk (all KT k-steps) of packed W into LDS via async copy.
// Per kt the slice is 4 ntiles * 512 elems = 4KB contiguous in BOTH src and dst,
// i.e. exactly 256 threads * 16B -> one async_cp16 per thread per kt.
template <int KT, int NTILES>
__device__ __forceinline__ void stage_chunk(const unsigned short* __restrict__ pW, int nt0,
                                            unsigned ldsBase, int tid) {
#pragma unroll
  for (int kt = 0; kt < KT; ++kt) {
    const unsigned short* src = pW + (size_t)(kt * NTILES + nt0) * 512 + tid * 8;
    unsigned ldsoff = ldsBase + (unsigned)(kt * 4 * 512 + tid * 8) * 2u;
    async_cp16(ldsoff, src);
  }
}

// Compute one staged chunk: this wave's M-tile (mw) x its 2 of the 4 staged
// n-tiles, accumulating over all KT k-steps. A from LDS (row-major bf16),
// B fragments from the LDS staging buffer (contiguous 32B per lane).
template <int KT, int LDA, bool RELU, bool F32OUT>
__device__ __forceinline__ void gemm_chunk(const unsigned short* __restrict__ sAct,
                                           const unsigned short* __restrict__ sB,
                                           int ntile0, int ntl0,
                                           unsigned short* __restrict__ dBf, int lddb,
                                           float* __restrict__ dF, int lddf,
                                           int mw, int lane) {
  const int l     = lane & 15;
  const int hi    = lane >> 4;
  const int mbase = mw * 16;
  const unsigned short* arow = sAct + (size_t)(mbase + l) * LDA;

  v8f acc0 = {}, acc1 = {};
#pragma unroll
  for (int kt = 0; kt < KT; ++kt) {
    const int kb = kt * 32;
    // A fragment (contiguous, matches packed B): lanes 0-15: K kb..kb+15,
    // lanes 16-31: K kb+16..kb+31.
    v8bf alo = *reinterpret_cast<const v8bf*>(arow + kb + hi * 16);
    v8bf ahi = *reinterpret_cast<const v8bf*>(arow + kb + hi * 16 + 8);
    v16bf a = __builtin_shufflevector(alo, ahi,
                0, 1, 2, 3, 4, 5, 6, 7, 8, 9, 10, 11, 12, 13, 14, 15);
    const unsigned short* bb = sB + ((size_t)(kt * 4 + ntl0) * 32 + lane) * 16;
    v16bf b0 = *reinterpret_cast<const v16bf*>(bb);
    v16bf b1 = *reinterpret_cast<const v16bf*>(bb + 32 * 16);
    acc0 = __builtin_amdgcn_wmma_f32_16x16x32_bf16(false, a, false, b0, (short)0, acc0, false, false);
    acc1 = __builtin_amdgcn_wmma_f32_16x16x32_bf16(false, a, false, b1, (short)0, acc1, false, false);
  }
  // C/D layout: VGPR i, lanes 0-15 -> (M = i, N = lane); lanes 16-31 -> (M = 8+i).
  const int nb = ntile0 * 16;
#pragma unroll
  for (int i = 0; i < 8; ++i) {
    int row = mbase + hi * 8 + i;
    float v0 = acc0[i], v1 = acc1[i];
    if (RELU) { v0 = fmaxf(v0, 0.f); v1 = fmaxf(v1, 0.f); }
    if (F32OUT) {
      float* dr = dF + (size_t)row * lddf + l;
      dr[nb] = v0; dr[nb + 16] = v1;
    } else {
      unsigned short* dr = dBf + (size_t)row * lddb + l;
      dr[nb] = f2bf(v0); dr[nb + 16] = f2bf(v1);
    }
  }
}

// Full GEMM phase with double-buffered async weight staging.
template <int KT, int NTILES, int LDA, bool RELU, bool F32OUT>
__device__ __forceinline__ void gemm_pipelined(const unsigned short* __restrict__ sAct,
                                               const unsigned short* __restrict__ pW,
                                               unsigned char* smem,
                                               unsigned short* __restrict__ dBf, int lddb,
                                               float* __restrict__ dF, int lddf,
                                               int tid, int wave, int lane) {
  constexpr int NCH = NTILES / 4;
  const int mw = wave >> 1;   // 0..3 : M-tile
  const int nw = wave & 1;    // 0..1 : which 2 of the 4 staged n-tiles
  const unsigned ldsB0 = (unsigned)(uintptr_t)(smem + LDS_SB0_OFF);
  const unsigned ldsB1 = (unsigned)(uintptr_t)(smem + LDS_SB1_OFF);
  const unsigned short* sB0 = (const unsigned short*)(smem + LDS_SB0_OFF);
  const unsigned short* sB1 = (const unsigned short*)(smem + LDS_SB1_OFF);

  stage_chunk<KT, NTILES>(pW, 0, ldsB0, tid);
  for (int c = 0; c < NCH; ++c) {
    const unsigned short* cur = (c & 1) ? sB1 : sB0;
    if (c + 1 < NCH) {
      stage_chunk<KT, NTILES>(pW, (c + 1) * 4, (c & 1) ? ldsB0 : ldsB1, tid);
      wait_async<KT>();   // batch c complete (in-order), batch c+1 may be in flight
    } else {
      wait_async<0>();
    }
    __syncthreads();      // staged chunk c visible to all waves
    gemm_chunk<KT, LDA, RELU, F32OUT>(sAct, cur, c * 4 + nw * 2, nw * 2,
                                      dBf, lddb, dF, lddf, mw, lane);
    __syncthreads();      // all waves done reading before buffer reuse
  }
}

__global__ void __launch_bounds__(256, 1)
pprgo_fused(const float* __restrict__ x,
            const int* __restrict__ topk_idx,
            const float* __restrict__ topk_w,
            const int* __restrict__ batch,
            const unsigned short* __restrict__ pW0,
            const unsigned short* __restrict__ pW1,
            const unsigned short* __restrict__ pW2,
            float* __restrict__ out) {
  extern __shared__ unsigned char smem[];
  unsigned short* sH1     = (unsigned short*)(smem + LDS_REGION0_OFF);  // 64x512 bf16
  float*          sLogits = (float*)(smem + LDS_REGION0_OFF);           // 64x192 f32 (aliases sH1)
  unsigned short* sA      = (unsigned short*)(smem + LDS_REGION1_OFF);  // 64x128 bf16
  unsigned short* sH2     = (unsigned short*)(smem + LDS_REGION1_OFF);  // 64x512 bf16 (aliases sA)
  int*   gidx = (int*)(smem + LDS_SMALL_OFF);          // [64]
  float* w_s  = (float*)(smem + LDS_SMALL_OFF + 256);  // [64]
  float* sAgg = (float*)(smem + LDS_SMALL_OFF + 512);  // [2*192]

  const int tid  = threadIdx.x;
  const int wave = tid >> 5;
  const int lane = tid & 31;
  const int wg   = blockIdx.x;

  // ---- phase 0: neighbor indices + PPR weights for this WG's 2 batch rows ----
  if (tid < MTILE) {
    int b    = wg * 2 + (tid >> 5);
    int k    = tid & (KNBR - 1);
    int node = batch[b];
    gidx[tid] = topk_idx[(size_t)node * KNBR + k];
    w_s[tid]  = topk_w[(size_t)node * KNBR + k];
  }
  __syncthreads();

  // ---- phase 1: gather x rows, f32 -> bf16 into sA (row-major 64x128) ----
  {
    int row = tid >> 2;
    int cb  = (tid & 3) * 32;
    const float4* src = (const float4*)(x + (size_t)gidx[row] * NUM_FEATURES + cb);
    unsigned* dst = (unsigned*)sA + ((size_t)row * NUM_FEATURES + cb) / 2;
#pragma unroll
    for (int i = 0; i < 8; ++i) {
      float4 v = src[i];
      dst[i * 2 + 0] = pack2bf(v.x, v.y);
      dst[i * 2 + 1] = pack2bf(v.z, v.w);
    }
  }
  __syncthreads();

  // ---- phase 2: H1 = relu(A @ W0)   [64x128]x[128x512] ----
  gemm_pipelined<4, 32, NUM_FEATURES, true, false>(sA, pW0, smem, sH1, HIDDEN,
                                                   nullptr, 0, tid, wave, lane);
  // ---- phase 3: H2 = relu(H1 @ W1)  [64x512]x[512x512] ----
  gemm_pipelined<16, 32, HIDDEN, true, false>(sH1, pW1, smem, sH2, HIDDEN,
                                              nullptr, 0, tid, wave, lane);
  // ---- phase 4: logits = H2 @ W2    [64x512]x[512x192] -> f32 ----
  gemm_pipelined<16, 12, HIDDEN, false, true>(sH2, pW2, smem, nullptr, 0,
                                              sLogits, NPAD, tid, wave, lane);

  // ---- phase 5: weighted aggregation over K=32 neighbors ----
  for (int e = tid; e < 2 * NPAD; e += 256) {
    int bi = e / NPAD, c = e % NPAD;
    const float* lg = sLogits + (size_t)(bi * KNBR) * NPAD + c;
    const float* wv = w_s + bi * KNBR;
    float s = 0.f;
#pragma unroll 8
    for (int k = 0; k < KNBR; ++k) s += wv[k] * lg[(size_t)k * NPAD];
    sAgg[e] = s;
  }
  __syncthreads();

  // ---- phase 6: log_softmax per batch row (one wave per row) ----
  if (wave < 2) {
    const float* rowp = sAgg + wave * NPAD;
    float m = -INFINITY;
    for (int c = lane; c < NUM_CLASSES; c += 32) m = fmaxf(m, rowp[c]);
#pragma unroll
    for (int o = 16; o > 0; o >>= 1) m = fmaxf(m, __shfl_xor(m, o, 32));
    float s = 0.f;
    for (int c = lane; c < NUM_CLASSES; c += 32) s += __expf(rowp[c] - m);
#pragma unroll
    for (int o = 16; o > 0; o >>= 1) s += __shfl_xor(s, o, 32);
    float lse = m + __logf(s);
    float* orow = out + (size_t)(wg * 2 + wave) * NUM_CLASSES;
    for (int c = lane; c < NUM_CLASSES; c += 32) orow[c] = rowp[c] - lse;
  }
}

extern "C" void kernel_launch(void* const* d_in, const int* in_sizes, int n_in,
                              void* d_out, int out_size, void* d_ws, size_t ws_size,
                              hipStream_t stream) {
  const float* x        = (const float*)d_in[0];
  const int*   topk_idx = (const int*)d_in[1];
  const float* topk_w   = (const float*)d_in[2];
  const int*   batch    = (const int*)d_in[3];
  const float* W0       = (const float*)d_in[4];
  const float* W1       = (const float*)d_in[5];
  const float* W2       = (const float*)d_in[6];

  // Workspace: packed bf16 weights in WMMA B-fragment order.
  unsigned short* pW0 = (unsigned short*)d_ws;        //  4*32*512 =  65536 elems (128KB)
  unsigned short* pW1 = pW0 + 4 * 32 * 512;           // 16*32*512 = 262144 elems (512KB)
  unsigned short* pW2 = pW1 + 16 * 32 * 512;          // 16*12*512 =  98304 elems (192KB)

  pprgo_pack_w<<<256, 256, 0, stream>>>(W0, pW0, NUM_FEATURES, HIDDEN, 4, 32);
  pprgo_pack_w<<<512, 256, 0, stream>>>(W1, pW1, HIDDEN, HIDDEN, 16, 32);
  pprgo_pack_w<<<256, 256, 0, stream>>>(W2, pW2, HIDDEN, NUM_CLASSES, 16, 12);

  (void)hipFuncSetAttribute((const void*)pprgo_fused,
                            hipFuncAttributeMaxDynamicSharedMemorySize, LDS_TOTAL);

  // 10000 batch rows / 2 per WG = 5000 workgroups; 256 threads = 8 wave32.
  pprgo_fused<<<5000, 256, LDS_TOTAL, stream>>>(x, topk_idx, topk_w, batch,
                                                pW0, pW1, pW2, (float*)d_out);
}
